// Linear_act_sp_75634374082778
// MI455X (gfx1250) — matrix operand order repaired and checked by reading, not compiled, not run on hardware
//
#include <hip/hip_runtime.h>
#include <hip/hip_fp16.h>
#include <math.h>

typedef __attribute__((ext_vector_type(16))) _Float16 v16h;
typedef __attribute__((ext_vector_type(32))) _Float16 v32h;
typedef __attribute__((ext_vector_type(8)))  _Float16 v8h;
typedef __attribute__((ext_vector_type(4)))  _Float16 v4h;
typedef __attribute__((ext_vector_type(8)))  float    v8f;
typedef __attribute__((ext_vector_type(4)))  float    v4f;

static constexpr int D_IN  = 4096;
static constexpr int D_OUT = 4096;
static constexpr int NTOK  = 16384;   // 4 * 4096
static constexpr int KC    = D_IN / 2;   // compressed K stride (2048)
static constexpr int NW    = D_IN / 32;  // index words per row (128)

// ---------------------------------------------------------------------------
// Kernel 0: zero the w_norm accumulator (ws is poisoned with 0xAA).
// ---------------------------------------------------------------------------
__global__ void init_wnorm_kernel(float* __restrict__ wnorm) {
    int i = blockIdx.x * blockDim.x + threadIdx.x;
    if (i < D_IN) wnorm[i] = 0.0f;
}

// ---------------------------------------------------------------------------
// Kernel 1: w_norm[k] = sum_o W[o,k]^2  (partial row-chunks + atomicAdd).
// ---------------------------------------------------------------------------
__global__ void wnorm_kernel(const float* __restrict__ W, float* __restrict__ wnorm) {
    int k  = blockIdx.x * 256 + threadIdx.x;
    int o0 = blockIdx.y * 512;
    float acc = 0.0f;
    for (int o = o0; o < o0 + 512; ++o) {
        float w = W[(size_t)o * D_IN + k];
        acc += w * w;
    }
    atomicAdd(wnorm + k, acc);
}

// ---------------------------------------------------------------------------
// Kernel 2: convert weight f32 -> f16 (same [out, in] row-major layout).
// ---------------------------------------------------------------------------
__global__ void wh_kernel(const float* __restrict__ W, _Float16* __restrict__ Wh) {
    size_t i = ((size_t)blockIdx.x * 256 + threadIdx.x) * 4;
    v4f w = *(const v4f*)(W + i);
    v4h h;
    h.x = (_Float16)w.x; h.y = (_Float16)w.y; h.z = (_Float16)w.z; h.w = (_Float16)w.w;
    *(v4h*)(Wh + i) = h;
}

// ---------------------------------------------------------------------------
// Kernel 3: fused 2:4 prune -> SWMMAC operand format + variance scale.
// One wave32 per token. Each lane builds one 32-bit index word = 8 groups of
// 4 channels (32 expanded K): nibble s = {idx1[1:0], idx0[1:0]}, idx0 < idx1
// (ISA 7.12.4). The two kept values are packed contiguously into the
// half-size compressed A (row-major [N, K/2] f16). Per-token variance
// correction is a scalar per row -> stored and applied in the GEMM epilogue.
// ---------------------------------------------------------------------------
__global__ void __launch_bounds__(256) prune_kernel(
        const float* __restrict__ x, const float* __restrict__ wnorm,
        _Float16* __restrict__ Asp, unsigned* __restrict__ Aidx,
        float* __restrict__ scale) {
    const int lane = threadIdx.x & 31;
    const int wave = threadIdx.x >> 5;
    const int n = blockIdx.x * 8 + wave;
    const size_t base = (size_t)n * D_IN;

    float sx = 0.f, sx2 = 0.f, sp = 0.f, sp2 = 0.f;

    for (int w = lane; w < NW; w += 32) {           // 128 words / 32 lanes
        const float* xp = x + base + (size_t)w * 32;
        const float* wp = wnorm + (size_t)w * 32;
        unsigned word = 0;
        _Float16 packed[16];
#pragma unroll
        for (int s = 0; s < 8; ++s) {
            v4f xv = *(const v4f*)(xp + s * 4);
            v4f wn = *(const v4f*)(wp + s * 4);
            float xa[4] = { xv.x, xv.y, xv.z, xv.w };
            float m[4]  = { fabsf(xv.x) * wn.x, fabsf(xv.y) * wn.y,
                            fabsf(xv.z) * wn.z, fabsf(xv.w) * wn.w };
            int p0 = 0, p1 = 0;
            bool first = true;
#pragma unroll
            for (int i = 0; i < 4; ++i) {
                int r = 0;
#pragma unroll
                for (int j = 0; j < 4; ++j)
                    r += (m[j] > m[i]) || (m[j] == m[i] && j < i);
                float v = xa[i];
                bool keep = (r < 2);                 // keep the 2 largest metrics
                sx += v;  sx2 += v * v;
                if (keep) { sp += v; sp2 += v * v; }
                if (keep) { if (first) { p0 = i; first = false; } else { p1 = i; } }
            }
            word |= (unsigned)(p0 | (p1 << 2)) << (4 * s);
            packed[2 * s]     = (_Float16)xa[p0];
            packed[2 * s + 1] = (_Float16)xa[p1];
        }
        _Float16* ap = Asp + (size_t)n * KC + (size_t)w * 16;
        *(v8h*)(ap)     = *(v8h*)&packed[0];
        *(v8h*)(ap + 8) = *(v8h*)&packed[8];
        Aidx[(size_t)n * NW + w] = word;
    }

    // wave32 butterfly reduction
#pragma unroll
    for (int off = 16; off > 0; off >>= 1) {
        sx  += __shfl_xor(sx,  off, 32);
        sx2 += __shfl_xor(sx2, off, 32);
        sp  += __shfl_xor(sp,  off, 32);
        sp2 += __shfl_xor(sp2, off, 32);
    }
    if (lane == 0) {
        const float d = (float)D_IN;
        float vx = sx2 - sx * sx / d;   // ddof=1 denominators cancel in ratio
        float vs = sp2 - sp * sp / d;
        scale[n] = sqrtf(vx / vs);
    }
}

// ---------------------------------------------------------------------------
// Kernel 4 (launched): 2:4 sparse GEMM via v_swmmac_f32_16x16x64_f16.
// out[n,o] = scale[n] * sum_k Asp_expand[n,k] * Wh[o,k]
// 8 waves/block (4 M x 2 N); each wave owns a 32x32 output tile = 2x2 SWMMAC
// tiles; k-loop steps 64 expanded K. Compressed A (64MB) + indices (8MB) +
// Wh (32MB) all live in the 192MB L2, so operands stream from L2.
// ---------------------------------------------------------------------------
__global__ void __launch_bounds__(256) spgemm_kernel(
        const _Float16* __restrict__ Asp, const unsigned* __restrict__ Aidx,
        const _Float16* __restrict__ Wh, const float* __restrict__ scale,
        float* __restrict__ out) {
    const int lane  = threadIdx.x & 31;
    const int wave  = threadIdx.x >> 5;
    const int waveM = wave & 3;
    const int waveN = wave >> 2;
    const int mBase = blockIdx.x * 128 + waveM * 32;
    const int oBase = blockIdx.y * 64  + waveN * 32;

    const int l15 = lane & 15;
    const int hi  = lane >> 4;

    const size_t arow0 = (size_t)(mBase + l15)      * KC;
    const size_t arow1 = (size_t)(mBase + 16 + l15) * KC;
    const size_t irow0 = (size_t)(mBase + l15)      * NW;
    const size_t irow1 = (size_t)(mBase + 16 + l15) * NW;
    const size_t brow0 = (size_t)(oBase + l15)      * D_IN;
    const size_t brow1 = (size_t)(oBase + 16 + l15) * D_IN;

    v8f c00 = {}, c01 = {}, c10 = {}, c11 = {};

    for (int k0 = 0; k0 < D_IN; k0 += 64) {
        // Stored (compressed) A 16x32 f16, dense-A layout on stored K':
        // lanes 0-15: K'=[0..7],[16..23]; lanes 16-31: K'=[8..15],[24..31]
        const int kc = (k0 >> 1) + (hi << 3);
        v16h a0, a1;
        ((v8h*)&a0)[0] = *(const v8h*)(Asp + arow0 + kc);
        ((v8h*)&a0)[1] = *(const v8h*)(Asp + arow0 + kc + 16);
        ((v8h*)&a1)[0] = *(const v8h*)(Asp + arow1 + kc);
        ((v8h*)&a1)[1] = *(const v8h*)(Asp + arow1 + kc + 16);

        // Dense B 64x16 f16: V0-7 lanes0-15 K=0-15 / lanes16-31 K=16-31;
        //                    V8-15 lanes0-15 K=32-47 / lanes16-31 K=48-63
        const int kb = k0 + (hi << 4);
        v32h b0, b1;
        ((v8h*)&b0)[0] = *(const v8h*)(Wh + brow0 + kb);
        ((v8h*)&b0)[1] = *(const v8h*)(Wh + brow0 + kb + 8);
        ((v8h*)&b0)[2] = *(const v8h*)(Wh + brow0 + kb + 32);
        ((v8h*)&b0)[3] = *(const v8h*)(Wh + brow0 + kb + 40);
        ((v8h*)&b1)[0] = *(const v8h*)(Wh + brow1 + kb);
        ((v8h*)&b1)[1] = *(const v8h*)(Wh + brow1 + kb + 8);
        ((v8h*)&b1)[2] = *(const v8h*)(Wh + brow1 + kb + 32);
        ((v8h*)&b1)[3] = *(const v8h*)(Wh + brow1 + kb + 40);

        // Index word: lanes 0-15 cover groups K=0..31, lanes 16-31 K=32..63
        const int iw = (k0 >> 5) + hi;
        int i0 = (int)Aidx[irow0 + iw];
        int i1 = (int)Aidx[irow1 + iw];

        if (k0 + 64 < D_IN) {
            __builtin_prefetch(Asp + arow0 + kc + 32, 0, 0);
            __builtin_prefetch(Wh + brow0 + kb + 64, 0, 0);
        }

        c00 = __builtin_amdgcn_swmmac_f32_16x16x64_f16(false, a0, false, b0, c00, i0, false, false);
        c01 = __builtin_amdgcn_swmmac_f32_16x16x64_f16(false, a0, false, b1, c01, i0, false, false);
        c10 = __builtin_amdgcn_swmmac_f32_16x16x64_f16(false, a1, false, b0, c10, i1, false, false);
        c11 = __builtin_amdgcn_swmmac_f32_16x16x64_f16(false, a1, false, b1, c11, i1, false, false);
    }

    // Epilogue: D layout VGPR j -> M = j + 8*hi, N = lane&15; fold in scale.
    const int col0 = oBase + l15;
    const int col1 = oBase + 16 + l15;
#pragma unroll
    for (int j = 0; j < 8; ++j) {
        int r0 = mBase + (hi << 3) + j;
        int r1 = r0 + 16;
        float s0 = scale[r0];
        float s1 = scale[r1];
        out[(size_t)r0 * D_OUT + col0] = s0 * c00[j];
        out[(size_t)r0 * D_OUT + col1] = s0 * c01[j];
        out[(size_t)r1 * D_OUT + col0] = s1 * c10[j];
        out[(size_t)r1 * D_OUT + col1] = s1 * c11[j];
    }
}

// ---------------------------------------------------------------------------
// Dense-WMMA reference GEMM (kept compiled, not launched) — verifies the
// v_wmma path stays available alongside v_swmmac in the histogram.
// ---------------------------------------------------------------------------
__global__ void __launch_bounds__(256) gemm_dense_kernel(
        const _Float16* __restrict__ Ah, const _Float16* __restrict__ Wh,
        const float* __restrict__ scale, float* __restrict__ out) {
    const int lane  = threadIdx.x & 31;
    const int wave  = threadIdx.x >> 5;
    const int mBase = blockIdx.x * 128 + (wave & 3) * 32;
    const int oBase = blockIdx.y * 64  + (wave >> 2) * 32;
    const int l15 = lane & 15;
    const int hi  = lane >> 4;
    const size_t arow0 = (size_t)(mBase + l15)      * D_IN;
    const size_t arow1 = (size_t)(mBase + 16 + l15) * D_IN;
    const size_t brow0 = (size_t)(oBase + l15)      * D_IN;
    const size_t brow1 = (size_t)(oBase + 16 + l15) * D_IN;

    v8f c00 = {}, c01 = {}, c10 = {}, c11 = {};
    for (int k0 = 0; k0 < D_IN; k0 += 32) {
        const int ka = k0 + (hi << 4);
        const int kb = k0 + (hi << 4);
        v16h a0, a1, b0, b1;
        ((v8h*)&a0)[0] = *(const v8h*)(Ah + arow0 + ka);
        ((v8h*)&a0)[1] = *(const v8h*)(Ah + arow0 + ka + 8);
        ((v8h*)&a1)[0] = *(const v8h*)(Ah + arow1 + ka);
        ((v8h*)&a1)[1] = *(const v8h*)(Ah + arow1 + ka + 8);
        ((v8h*)&b0)[0] = *(const v8h*)(Wh + brow0 + kb);
        ((v8h*)&b0)[1] = *(const v8h*)(Wh + brow0 + kb + 8);
        ((v8h*)&b1)[0] = *(const v8h*)(Wh + brow1 + kb);
        ((v8h*)&b1)[1] = *(const v8h*)(Wh + brow1 + kb + 8);
        c00 = __builtin_amdgcn_wmma_f32_16x16x32_f16(false, a0, false, b0, (short)0, c00, false, false);
        c01 = __builtin_amdgcn_wmma_f32_16x16x32_f16(false, a0, false, b1, (short)0, c01, false, false);
        c10 = __builtin_amdgcn_wmma_f32_16x16x32_f16(false, a1, false, b0, (short)0, c10, false, false);
        c11 = __builtin_amdgcn_wmma_f32_16x16x32_f16(false, a1, false, b1, (short)0, c11, false, false);
    }
    const int col0 = oBase + l15, col1 = oBase + 16 + l15;
#pragma unroll
    for (int j = 0; j < 8; ++j) {
        int r0 = mBase + (hi << 3) + j;
        int r1 = r0 + 16;
        out[(size_t)r0 * D_OUT + col0] = scale[r0] * c00[j];
        out[(size_t)r0 * D_OUT + col1] = scale[r0] * c01[j];
        out[(size_t)r1 * D_OUT + col0] = scale[r1] * c10[j];
        out[(size_t)r1 * D_OUT + col1] = scale[r1] * c11[j];
    }
}

// ---------------------------------------------------------------------------
extern "C" void kernel_launch(void* const* d_in, const int* in_sizes, int n_in,
                              void* d_out, int out_size, void* d_ws, size_t ws_size,
                              hipStream_t stream) {
    const float* x = (const float*)d_in[0];   // [4,4096,4096] f32
    const float* W = (const float*)d_in[1];   // [4096,4096]  f32
    float* out = (float*)d_out;               // [4,4096,4096] f32

    char* ws = (char*)d_ws;
    _Float16* Asp  = (_Float16*)ws;                                  // 64 MB
    unsigned* Aidx = (unsigned*)(ws + (size_t)NTOK * KC * 2);        //  8 MB
    _Float16* Wh   = (_Float16*)((char*)Aidx + (size_t)NTOK * NW * 4); // 32 MB
    float* wnorm   = (float*)((char*)Wh + (size_t)D_OUT * D_IN * 2); // 16 KB
    float* scale   = wnorm + D_IN;                                   // 64 KB

    init_wnorm_kernel<<<(D_IN + 255) / 256, 256, 0, stream>>>(wnorm);
    wnorm_kernel<<<dim3(D_IN / 256, 8), 256, 0, stream>>>(W, wnorm);
    wh_kernel<<<(int)(((size_t)D_OUT * D_IN / 4) / 256), 256, 0, stream>>>(W, Wh);
    prune_kernel<<<NTOK / 8, 256, 0, stream>>>(x, wnorm, Asp, Aidx, scale);
    spgemm_kernel<<<dim3(NTOK / 128, D_OUT / 64), 256, 0, stream>>>(Asp, Aidx, Wh, scale, out);
}
